// GeniePathConv_21930103014154
// MI455X (gfx1250) — matrix operand order repaired and checked by reading, not compiled, not run on hardware
//
#include <hip/hip_runtime.h>
#include <math.h>

#define N_NODES 50000
#define N_EDGES 800000
#define IN_DIM  256
#define HID     96
#define OUT_DIM 128
#define GATES   512            // 4*OUT
#define K_LSTM  224            // HID + OUT
#define MT      (N_NODES / 16) // 3125 row tiles (exact)

typedef __attribute__((ext_vector_type(16))) _Float16 v16h;
typedef __attribute__((ext_vector_type(8)))  float    v8f;

// ---- WMMA fragment helpers (CDNA5 ISA 7.12.2, 16-bit operands) ----
// A 16x32: lane<16 -> row M=lane, K in {0..7,16..23}; lane>=16 -> same row, +8.
// Per lane the 16 K-indices are two contiguous runs of 8: [8h..8h+7], [16+8h..23+8h].
// B 32x16: lane -> col N=lane&15, K = (e&7) + 8*(lane>>4) + 16*(e>>3) (same order as A).
__device__ __forceinline__ v16h load_a_frag(const float* __restrict__ run0,
                                            const float* __restrict__ run1) {
    float4 q0 = ((const float4*)run0)[0];
    float4 q1 = ((const float4*)run0)[1];
    float4 q2 = ((const float4*)run1)[0];
    float4 q3 = ((const float4*)run1)[1];
    v16h a;
    a[0]  = (_Float16)q0.x; a[1]  = (_Float16)q0.y; a[2]  = (_Float16)q0.z; a[3]  = (_Float16)q0.w;
    a[4]  = (_Float16)q1.x; a[5]  = (_Float16)q1.y; a[6]  = (_Float16)q1.z; a[7]  = (_Float16)q1.w;
    a[8]  = (_Float16)q2.x; a[9]  = (_Float16)q2.y; a[10] = (_Float16)q2.z; a[11] = (_Float16)q2.w;
    a[12] = (_Float16)q3.x; a[13] = (_Float16)q3.y; a[14] = (_Float16)q3.z; a[15] = (_Float16)q3.w;
    return a;
}

__device__ __forceinline__ float sigm(float x) { return 1.0f / (1.0f + expf(-x)); }

// ordered-uint mapping so atomicMax(uint) == float max (handles negatives)
__device__ __forceinline__ unsigned f2ord(float f) {
    unsigned b = __float_as_uint(f);
    return (b & 0x80000000u) ? ~b : (b | 0x80000000u);
}
__device__ __forceinline__ float ord2f(unsigned u) {
    return (u & 0x80000000u) ? __uint_as_float(u ^ 0x80000000u) : __uint_as_float(~u);
}

// ---------------- init scratch ----------------
__global__ void k_init(float* rst, float* esum, unsigned* emaxu) {
    int i = blockIdx.x * blockDim.x + threadIdx.x;
    if (i < N_NODES * HID) rst[i] = 0.0f;
    if (i < N_NODES) { esum[i] = 0.0f; emaxu[i] = 0u; } // 0 < ord(-inf): safe floor
}

// ---------------- feat = x @ W_fc  (WMMA f16 -> f32) ----------------
// B staged in LDS in FRAGMENT-MAJOR layout: frag f = ko*6 + t, then [lane][e]
// contiguous 16 halves per lane -> single v16h (2x ds_load_b128) per fragment.
#define FC_FRAGS (8 * 6) // ko-tiles * n-tiles
__global__ void k_fc_gemm(const float* __restrict__ x, const float* __restrict__ Wfc,
                          float* __restrict__ feat) {
    __shared__ v16h sW[FC_FRAGS * 32]; // 48 KB
    {
        _Float16* sWh = (_Float16*)sW;
        for (int i = threadIdx.x; i < FC_FRAGS * 512; i += blockDim.x) {
            int f = i >> 9, rem = i & 511;
            int l = rem >> 4, e = rem & 15;
            int ko = f / 6, t = f - 6 * ko;
            int k = ko * 32 + (e & 7) + 8 * (l >> 4) + 16 * (e >> 3);
            int n = t * 16 + (l & 15);
            sWh[i] = (_Float16)Wfc[k * HID + n];
        }
    }
    __syncthreads();

    const int wave = threadIdx.x >> 5;
    const int lane = threadIdx.x & 31;
    const int half = lane >> 4;
    const int lrow = lane & 15;
    const int wstride = (blockDim.x >> 5) * gridDim.x;

    for (int m = blockIdx.x * (blockDim.x >> 5) + wave; m < MT; m += wstride) {
        v8f acc[6];
#pragma unroll
        for (int t = 0; t < 6; ++t) acc[t] = (v8f){};
        const int row = m * 16 + lrow;
#pragma unroll 2
        for (int ko = 0; ko < IN_DIM / 32; ++ko) {
            const float* base = x + (size_t)row * IN_DIM + ko * 32 + 8 * half;
            v16h a = load_a_frag(base, base + 16);
#pragma unroll
            for (int t = 0; t < 6; ++t) {
                v16h b = sW[(ko * 6 + t) * 32 + lane];
                acc[t] = __builtin_amdgcn_wmma_f32_16x16x32_f16(
                    false, a, false, b, (short)0, acc[t], false, false);
            }
        }
#pragma unroll
        for (int t = 0; t < 6; ++t)
#pragma unroll
            for (int r = 0; r < 8; ++r)
                feat[(size_t)(m * 16 + r + 8 * half) * HID + t * 16 + lrow] = acc[t][r];
    }
}

// ---------------- el/er attention dots ----------------
__global__ void k_el_er(const float* __restrict__ feat, const float* __restrict__ al,
                        const float* __restrict__ ar, float* el, float* er) {
    int n = blockIdx.x * blockDim.x + threadIdx.x;
    if (n >= N_NODES) return;
    float a = 0.0f, b = 0.0f;
    const float4* f4 = (const float4*)(feat + (size_t)n * HID);
#pragma unroll 4
    for (int d = 0; d < HID / 4; ++d) {
        float4 f = f4[d];
        float4 l = ((const float4*)al)[d];
        float4 r = ((const float4*)ar)[d];
        a += f.x * l.x + f.y * l.y + f.z * l.z + f.w * l.w;
        b += f.x * r.x + f.y * r.y + f.z * r.z + f.w * r.w;
    }
    el[n] = a; er[n] = b;
}

// ---------------- edge leaky-relu + segment max ----------------
__global__ void k_edge_logits(const float* __restrict__ el, const float* __restrict__ er,
                              const int* __restrict__ src, const int* __restrict__ dst,
                              float* ebuf, unsigned* emaxu) {
    int i = blockIdx.x * blockDim.x + threadIdx.x;
    if (i >= N_EDGES) return;
    float v = el[src[i]] + er[dst[i]];
    v = v > 0.0f ? v : 0.2f * v;
    ebuf[i] = v;
    atomicMax(&emaxu[dst[i]], f2ord(v));
}

// ---------------- exp + segment sum ----------------
__global__ void k_edge_exp(const float* __restrict__ ebuf, const unsigned* __restrict__ emaxu,
                           const int* __restrict__ dst, float* exbuf, float* esum) {
    int i = blockIdx.x * blockDim.x + threadIdx.x;
    if (i >= N_EDGES) return;
    int d = dst[i];
    float ex = expf(ebuf[i] - ord2f(emaxu[d]));
    exbuf[i] = ex;
    atomicAdd(&esum[d], ex);
}

// ---------------- alpha * feat[src] scatter-add (wave per edge) ----------------
__global__ void k_edge_agg(const float* __restrict__ exbuf, const float* __restrict__ esum,
                           const float* __restrict__ feat, const int* __restrict__ src,
                           const int* __restrict__ dst, float* rst) {
    int edge = blockIdx.x * (blockDim.x >> 5) + (threadIdx.x >> 5);
    if (edge >= N_EDGES) return;
    int lane = threadIdx.x & 31;
    int s = src[edge], dd = dst[edge];
    float alpha = exbuf[edge] / esum[dd];
#pragma unroll
    for (int t = 0; t < 3; ++t) {
        int d = lane + 32 * t;
        atomicAdd(&rst[(size_t)dd * HID + d], alpha * feat[(size_t)s * HID + d]);
    }
}

// ---------------- xmid = tanh(rst + bias) ----------------
__global__ void k_tanh(const float* __restrict__ rst, const float* __restrict__ bias,
                       float* xmid) {
    int i = blockIdx.x * blockDim.x + threadIdx.x; // float4 index
    if (i >= (N_NODES * HID) / 4) return;
    float4 r = ((const float4*)rst)[i];
    float4 b = ((const float4*)bias)[(i * 4) % HID / 4];
    float4 o;
    o.x = tanhf(r.x + b.x); o.y = tanhf(r.y + b.y);
    o.z = tanhf(r.z + b.z); o.w = tanhf(r.w + b.w);
    ((float4*)xmid)[i] = o;
}

// ---------------- fused LSTM: gates GEMM (WMMA) + nonlinearity epilogue ----------------
// A = [xmid | h0] (N x 224), B = [W_ih.T ; W_hh.T] (224 x 512), fragment-major in LDS.
// Wave w owns the 16-col slice (w*16..) of ALL four gates -> full epilogue in-register.
#define LSTM_FRAGS (7 * 32) // ko-tiles * (4 gates * 8 jtiles)
__global__ void k_lstm(const float* __restrict__ xmid, const float* __restrict__ h0,
                       const float* __restrict__ c0,
                       const float* __restrict__ W_ih, const float* __restrict__ W_hh,
                       const float* __restrict__ b_ih, const float* __restrict__ b_hh,
                       float* __restrict__ out) {
    extern __shared__ v16h sB[]; // 224 KB (<= 320 KB WGP LDS)
    {
        _Float16* sBh = (_Float16*)sB;
        for (int i = threadIdx.x; i < LSTM_FRAGS * 512; i += blockDim.x) {
            int f = i >> 9, rem = i & 511;
            int l = rem >> 4, e = rem & 15;
            int ko = f >> 5, t = f & 31; // t = g*8 + jtile
            int k = ko * 32 + (e & 7) + 8 * (l >> 4) + 16 * (e >> 3);
            int n = t * 16 + (l & 15);
            float v = (k < HID) ? W_ih[(size_t)n * HID + k]
                                : W_hh[(size_t)n * OUT_DIM + (k - HID)];
            sBh[i] = (_Float16)v;
        }
    }
    __syncthreads();

    const int wave = threadIdx.x >> 5; // jtile 0..7
    const int lane = threadIdx.x & 31;
    const int half = lane >> 4;
    const int lrow = lane & 15;
    const int jcol = wave * 16 + lrow; // column within each gate, 0..127

    float* h1a = out;
    float* h1b = out + (size_t)N_NODES * OUT_DIM;
    float* c1o = out + 2 * (size_t)N_NODES * OUT_DIM;

    const float bi = b_ih[0 * OUT_DIM + jcol] + b_hh[0 * OUT_DIM + jcol];
    const float bf = b_ih[1 * OUT_DIM + jcol] + b_hh[1 * OUT_DIM + jcol];
    const float bg = b_ih[2 * OUT_DIM + jcol] + b_hh[2 * OUT_DIM + jcol];
    const float bo = b_ih[3 * OUT_DIM + jcol] + b_hh[3 * OUT_DIM + jcol];

    for (int m = blockIdx.x; m < MT; m += gridDim.x) {
        v8f acc[4];
#pragma unroll
        for (int g = 0; g < 4; ++g) acc[g] = (v8f){};
        const int row = m * 16 + lrow;
#pragma unroll 1
        for (int ko = 0; ko < K_LSTM / 32; ++ko) {
            // two 8-float runs; 96 is a multiple of 8 so a run never straddles sources
            int s0 = ko * 32 + 8 * half;
            int s1 = s0 + 16;
            const float* r0 = (s0 < HID) ? xmid + (size_t)row * HID + s0
                                         : h0 + (size_t)row * OUT_DIM + (s0 - HID);
            const float* r1 = (s1 < HID) ? xmid + (size_t)row * HID + s1
                                         : h0 + (size_t)row * OUT_DIM + (s1 - HID);
            v16h a = load_a_frag(r0, r1);
#pragma unroll
            for (int g = 0; g < 4; ++g) {
                v16h b = sB[(ko * 32 + g * 8 + wave) * 32 + lane];
                acc[g] = __builtin_amdgcn_wmma_f32_16x16x32_f16(
                    false, a, false, b, (short)0, acc[g], false, false);
            }
        }
#pragma unroll
        for (int r = 0; r < 8; ++r) {
            int grow = m * 16 + r + 8 * half;
            float iv = acc[0][r] + bi;
            float fv = acc[1][r] + bf;
            float gv = acc[2][r] + bg;
            float ov = acc[3][r] + bo;
            size_t oidx = (size_t)grow * OUT_DIM + jcol;
            float c1 = sigm(fv) * c0[oidx] + sigm(iv) * tanhf(gv);
            float h1 = sigm(ov) * tanhf(c1);
            h1a[oidx] = h1;
            h1b[oidx] = h1;
            c1o[oidx] = c1;
        }
    }
}

extern "C" void kernel_launch(void* const* d_in, const int* in_sizes, int n_in,
                              void* d_out, int out_size, void* d_ws, size_t ws_size,
                              hipStream_t stream) {
    const float* x      = (const float*)d_in[0];
    const float* W_fc   = (const float*)d_in[1];
    const float* attn_l = (const float*)d_in[2];
    const float* attn_r = (const float*)d_in[3];
    const float* bias   = (const float*)d_in[4];
    const float* W_ih   = (const float*)d_in[5];
    const float* W_hh   = (const float*)d_in[6];
    const float* b_ih   = (const float*)d_in[7];
    const float* b_hh   = (const float*)d_in[8];
    const float* h0     = (const float*)d_in[9];
    const float* c0     = (const float*)d_in[10];
    const int*   src    = (const int*)d_in[11];
    const int*   dst    = (const int*)d_in[12];
    float* out = (float*)d_out;

    // workspace carve-up (~65 MB of fp32)
    float* ws = (float*)d_ws;
    size_t off = 0;
    float*    feat  = ws + off; off += (size_t)N_NODES * HID;
    float*    el    = ws + off; off += N_NODES;
    float*    er    = ws + off; off += N_NODES;
    unsigned* emaxu = (unsigned*)(ws + off); off += N_NODES;
    float*    esum  = ws + off; off += N_NODES;
    float*    ebuf  = ws + off; off += N_EDGES;
    float*    exbuf = ws + off; off += N_EDGES;
    float*    rst   = ws + off; off += (size_t)N_NODES * HID;
    float*    xmid  = ws + off; off += (size_t)N_NODES * HID;
    (void)ws_size; (void)in_sizes; (void)n_in; (void)out_size;

    const int ELTS = N_NODES * HID;
    k_init<<<(ELTS + 255) / 256, 256, 0, stream>>>(rst, esum, emaxu);
    k_fc_gemm<<<391, 256, 0, stream>>>(x, W_fc, feat);
    k_el_er<<<(N_NODES + 255) / 256, 256, 0, stream>>>(feat, attn_l, attn_r, el, er);
    k_edge_logits<<<(N_EDGES + 255) / 256, 256, 0, stream>>>(el, er, src, dst, ebuf, emaxu);
    k_edge_exp<<<(N_EDGES + 255) / 256, 256, 0, stream>>>(ebuf, emaxu, dst, exbuf, esum);
    k_edge_agg<<<(N_EDGES + 7) / 8, 256, 0, stream>>>(exbuf, esum, feat, src, dst, rst);
    k_tanh<<<(ELTS / 4 + 255) / 256, 256, 0, stream>>>(rst, bias, xmid);
    k_lstm<<<256, 256, (size_t)LSTM_FRAGS * 512 * sizeof(_Float16), stream>>>(
        xmid, h0, c0, W_ih, W_hh, b_ih, b_hh, out);
}
